// GraphNetwork_51891794871089
// MI455X (gfx1250) — matrix operand order repaired and checked by reading, not compiled, not run on hardware
//
#include <hip/hip_runtime.h>
#include <hip/hip_bf16.h>

// ---------------------------------------------------------------------------
// GraphNetwork block for MI455X (gfx1250): bf16 WMMA MLPs + scatter/segment ops
// Wave->tile mapping: each wave owns BOTH M-tiles so every weight (B) fragment
// is loaded from L2 exactly once per block; extra A fragments come from LDS.
// ---------------------------------------------------------------------------

typedef __bf16 bf16_t;
typedef __attribute__((ext_vector_type(16))) __bf16 v16bf;
typedef __attribute__((ext_vector_type(8)))  __bf16 v8bf;
typedef __attribute__((ext_vector_type(4)))  __bf16 v4bf;
typedef __attribute__((ext_vector_type(8)))  float  v8f;

#define N_NODES 40000
#define N_EDGES 640000
#define N_GRAPH 8
#define DIM     128
#define HID     256
#define MT      32            // rows (edges/nodes) per block
#define HSTRIDE (HID + 8)     // hidden LDS stride (bf16 elems), 16B-aligned rows

__device__ __forceinline__ void atomAddF(float* p, float v) {
  (void)unsafeAtomicAdd(p, v);   // global_atomic_add_f32 / ds_add_f32, no return
}

// A fragment (16x32 bf16, MxK) from LDS. Per ISA layout: lanes 0-15 hold rows
// M=0..15 with K = 0..7 (v0-3) and 16..23 (v4-7); lanes 16-31 hold K=8..15 and
// K=24..31 for the same rows. Two 16-byte ds loads per lane.
__device__ __forceinline__ v16bf frag_a_lds(const bf16_t* base, int stride,
                                            int m0, int k0, int lane) {
  const int m  = m0 + (lane & 15);
  const int kh = (lane >> 4) << 3;                 // 0 or 8
  union { v16bf v; v8bf h[2]; } u;
  u.h[0] = *(const v8bf*)(base + m * stride + k0 + kh);
  u.h[1] = *(const v8bf*)(base + m * stride + k0 + 16 + kh);
  return u.v;
}

// B fragment (32x16 bf16, KxN) from global weights stored transposed [N][K]:
// lane holds column n = lane&15, K = (lane>>4)*8 .. +7 and 16+(lane>>4)*8 .. +7.
__device__ __forceinline__ v16bf frag_b_glb(const bf16_t* __restrict__ wT,
                                            int K, int n0, int k0, int lane) {
  const int n = n0 + (lane & 15);
  const int k = k0 + ((lane >> 4) << 3);           // 0 or 8
  union { v16bf v; v8bf h[2]; } u;
  u.h[0] = *(const v8bf*)(wT + (size_t)n * K + k);
  u.h[1] = *(const v8bf*)(wT + (size_t)n * K + k + 16);
  return u.v;
}

__device__ __forceinline__ v8f wmma_bf16(v16bf a, v16bf b, v8f c) {
  return __builtin_amdgcn_wmma_f32_16x16x32_bf16(false, a, false, b,
                                                 (short)0, c, false, false);
}

// ---------------------------------------------------------------------------
__global__ void zero_kernel(float* __restrict__ p, long n) {
  long i = (long)blockIdx.x * blockDim.x + threadIdx.x;
  const long stride = (long)gridDim.x * blockDim.x;
  for (; i < n; i += stride) p[i] = 0.f;
}

// src is [K][N] fp32 row-major; dst is [N][K] bf16 (i.e. W^T).
__global__ void transpose_bf16_kernel(const float* __restrict__ src,
                                      bf16_t* __restrict__ dst, int K, int N) {
  const int i = blockIdx.x * blockDim.x + threadIdx.x;
  if (i < K * N) {
    const int k = i / N, n = i % N;
    dst[(size_t)n * K + k] = (bf16_t)src[i];
  }
}

__global__ __launch_bounds__(256) void count_kernel(
    const int* __restrict__ nidx, const int* __restrict__ eidx,
    float* __restrict__ ncnt, float* __restrict__ ecnt) {
  __shared__ float loc[16];
  const int tid = threadIdx.x;
  if (tid < 16) loc[tid] = 0.f;
  __syncthreads();
  const int i = blockIdx.x * 256 + tid;
  if (i < N_NODES) atomAddF(&loc[nidx[i] & 7], 1.f);
  if (i < N_EDGES) atomAddF(&loc[8 + (eidx[i] & 7)], 1.f);
  __syncthreads();
  if (tid < 8)            { if (loc[tid] != 0.f) atomAddF(&ncnt[tid], loc[tid]); }
  else if (tid < 16)      { if (loc[tid] != 0.f) atomAddF(&ecnt[tid - 8], loc[tid]); }
}

// ---------------------------------------------------------------------------
// Edge MLP: [E,512] -> relu([.,512]x[512,256]+b) x [256,128] + b
// Also scatter-adds result into agg[col] and per-graph edge sums.
__global__ __launch_bounds__(256) void edge_mlp_kernel(
    const float* __restrict__ node_attr, const float* __restrict__ edge_attr,
    const float* __restrict__ uattr,
    const int* __restrict__ erow, const int* __restrict__ ecol,
    const int* __restrict__ egid,
    const bf16_t* __restrict__ W1T, const float* __restrict__ b1,
    const bf16_t* __restrict__ W2T, const float* __restrict__ b2,
    float* __restrict__ edge_out, float* __restrict__ agg,
    float* __restrict__ edgesum) {
  __shared__ __align__(16) bf16_t As[MT * 512];        // 32 KB
  __shared__ __align__(16) bf16_t Hsm[MT * HSTRIDE];   // 16.5 KB
  __shared__ float pSum[N_GRAPH * DIM];                // 4 KB

  const int tid  = threadIdx.x;
  const int lane = tid & 31;
  const int wave = tid >> 5;
  const int e0   = blockIdx.x * MT;

  for (int i = tid; i < N_GRAPH * DIM; i += 256) pSum[i] = 0.f;

  // ---- gather [node[row] | node[col] | edge | u[g]] -> bf16 LDS tile ----
  {
    const int r   = tid >> 3;       // 0..31 (edge row within tile)
    const int sub = tid & 7;        // 8 threads per row
    const int e   = e0 + r;
    const bool valid = (e < N_EDGES);
    int rw = 0, cl = 0, g = 0;
    if (valid) { rw = erow[e]; cl = ecol[e]; g = egid[e]; }
#pragma unroll
    for (int ii = 0; ii < 16; ++ii) {
      const int i = (sub << 2) + (ii << 5);   // 4 elems per iter, stride 32
      float4 v = make_float4(0.f, 0.f, 0.f, 0.f);
      if (valid) {
        const int c = i >> 7, off = i & 127;
        const float* sp = (c == 0) ? node_attr + (size_t)rw * DIM
                        : (c == 1) ? node_attr + (size_t)cl * DIM
                        : (c == 2) ? edge_attr + (size_t)e  * DIM
                                   : uattr     + (size_t)g  * DIM;
        v = *(const float4*)(sp + off);
      }
      v4bf pk = { (bf16_t)v.x, (bf16_t)v.y, (bf16_t)v.z, (bf16_t)v.w };
      *(v4bf*)(As + r * 512 + i) = pk;
    }
  }
  __syncthreads();

  // ---- GEMM1: K=512. Wave w owns both M-tiles x 2 N-tiles {2w, 2w+1}.
  //      Each W1 fragment is loaded exactly once per block. ----
  {
    const int nb = wave << 1;
    v8f acc[2][2] = {};                        // [m-tile][n-tile]
    for (int k = 0; k < 512; k += 32) {
      v16bf a0 = frag_a_lds(As, 512, 0,  k, lane);
      v16bf a1 = frag_a_lds(As, 512, 16, k, lane);
#pragma unroll
      for (int j = 0; j < 2; ++j) {
        v16bf b = frag_b_glb(W1T, 512, (nb + j) * 16, k, lane);
        acc[0][j] = wmma_bf16(a0, b, acc[0][j]);
        acc[1][j] = wmma_bf16(a1, b, acc[1][j]);
      }
    }
#pragma unroll
    for (int j = 0; j < 2; ++j) {
      const int n = (nb + j) * 16 + (lane & 15);
      const float bias = b1[n];
#pragma unroll
      for (int jm = 0; jm < 2; ++jm) {
#pragma unroll
        for (int r2 = 0; r2 < 8; ++r2) {
          const int m = jm * 16 + r2 + ((lane >> 4) << 3);
          Hsm[m * HSTRIDE + n] = (bf16_t)fmaxf(acc[jm][j][r2] + bias, 0.f);
        }
      }
    }
  }
  __syncthreads();

  // ---- GEMM2: K=256. Wave w owns both M-tiles x 1 N-tile (w). ----
  {
    const int n0t = wave * 16;
    v8f acc[2] = {};                           // [m-tile]
    for (int k = 0; k < 256; k += 32) {
      v16bf a0 = frag_a_lds(Hsm, HSTRIDE, 0,  k, lane);
      v16bf a1 = frag_a_lds(Hsm, HSTRIDE, 16, k, lane);
      v16bf b  = frag_b_glb(W2T, 256, n0t, k, lane);
      acc[0] = wmma_bf16(a0, b, acc[0]);
      acc[1] = wmma_bf16(a1, b, acc[1]);
    }
    const int n = n0t + (lane & 15);
    const float bias = b2[n];
#pragma unroll
    for (int jm = 0; jm < 2; ++jm) {
#pragma unroll
      for (int r2 = 0; r2 < 8; ++r2) {
        const int m = jm * 16 + r2 + ((lane >> 4) << 3);
        const int e = e0 + m;
        if (e < N_EDGES) {
          const int cl = ecol[e];
          const int g  = egid[e] & 7;
          const float v = acc[jm][r2] + bias;
          edge_out[(size_t)e * DIM + n] = v;
          atomAddF(&agg[(size_t)cl * DIM + n], v);     // segment_sum by col
          atomAddF(&pSum[g * DIM + n], v);             // per-graph edge sum
        }
      }
    }
  }
  __syncthreads();
  for (int i = tid; i < N_GRAPH * DIM; i += 256)
    if (pSum[i] != 0.f) atomAddF(&edgesum[i], pSum[i]);
}

// ---------------------------------------------------------------------------
// Node MLP: [N,384] -> relu(x W1 + b1) W2 + b2 ; also per-graph node sums.
__global__ __launch_bounds__(256) void node_mlp_kernel(
    const float* __restrict__ node_attr, const float* __restrict__ agg,
    const float* __restrict__ uattr, const int* __restrict__ ngid,
    const bf16_t* __restrict__ W1T, const float* __restrict__ b1,
    const bf16_t* __restrict__ W2T, const float* __restrict__ b2,
    float* __restrict__ node_out, float* __restrict__ nodesum) {
  __shared__ __align__(16) bf16_t As[MT * 384];        // 24 KB
  __shared__ __align__(16) bf16_t Hsm[MT * HSTRIDE];   // 16.5 KB
  __shared__ float pSum[N_GRAPH * DIM];                // 4 KB

  const int tid  = threadIdx.x;
  const int lane = tid & 31;
  const int wave = tid >> 5;
  const int n0   = blockIdx.x * MT;

  for (int i = tid; i < N_GRAPH * DIM; i += 256) pSum[i] = 0.f;

  // ---- gather [node | agg | u[g]] -> bf16 LDS tile ----
  {
    const int r   = tid >> 3;
    const int sub = tid & 7;
    const int nd  = n0 + r;
    const bool valid = (nd < N_NODES);
    int g = 0;
    if (valid) g = ngid[nd] & 7;
#pragma unroll
    for (int ii = 0; ii < 12; ++ii) {
      const int i = (sub << 2) + (ii << 5);
      float4 v = make_float4(0.f, 0.f, 0.f, 0.f);
      if (valid) {
        const int c = i >> 7, off = i & 127;
        const float* sp = (c == 0) ? node_attr + (size_t)nd * DIM
                        : (c == 1) ? agg       + (size_t)nd * DIM
                                   : uattr     + (size_t)g  * DIM;
        v = *(const float4*)(sp + off);
      }
      v4bf pk = { (bf16_t)v.x, (bf16_t)v.y, (bf16_t)v.z, (bf16_t)v.w };
      *(v4bf*)(As + r * 384 + i) = pk;
    }
  }
  __syncthreads();

  // ---- GEMM1: K=384. Wave w owns both M-tiles x 2 N-tiles. ----
  {
    const int nb = wave << 1;
    v8f acc[2][2] = {};
    for (int k = 0; k < 384; k += 32) {
      v16bf a0 = frag_a_lds(As, 384, 0,  k, lane);
      v16bf a1 = frag_a_lds(As, 384, 16, k, lane);
#pragma unroll
      for (int j = 0; j < 2; ++j) {
        v16bf b = frag_b_glb(W1T, 384, (nb + j) * 16, k, lane);
        acc[0][j] = wmma_bf16(a0, b, acc[0][j]);
        acc[1][j] = wmma_bf16(a1, b, acc[1][j]);
      }
    }
#pragma unroll
    for (int j = 0; j < 2; ++j) {
      const int n = (nb + j) * 16 + (lane & 15);
      const float bias = b1[n];
#pragma unroll
      for (int jm = 0; jm < 2; ++jm) {
#pragma unroll
        for (int r2 = 0; r2 < 8; ++r2) {
          const int m = jm * 16 + r2 + ((lane >> 4) << 3);
          Hsm[m * HSTRIDE + n] = (bf16_t)fmaxf(acc[jm][j][r2] + bias, 0.f);
        }
      }
    }
  }
  __syncthreads();

  // ---- GEMM2: K=256. Wave w owns both M-tiles x 1 N-tile. ----
  {
    const int n0t = wave * 16;
    v8f acc[2] = {};
    for (int k = 0; k < 256; k += 32) {
      v16bf a0 = frag_a_lds(Hsm, HSTRIDE, 0,  k, lane);
      v16bf a1 = frag_a_lds(Hsm, HSTRIDE, 16, k, lane);
      v16bf b  = frag_b_glb(W2T, 256, n0t, k, lane);
      acc[0] = wmma_bf16(a0, b, acc[0]);
      acc[1] = wmma_bf16(a1, b, acc[1]);
    }
    const int n = n0t + (lane & 15);
    const float bias = b2[n];
#pragma unroll
    for (int jm = 0; jm < 2; ++jm) {
#pragma unroll
      for (int r2 = 0; r2 < 8; ++r2) {
        const int m = jm * 16 + r2 + ((lane >> 4) << 3);
        const int nd = n0 + m;
        if (nd < N_NODES) {
          const int g = ngid[nd] & 7;
          const float v = acc[jm][r2] + bias;
          node_out[(size_t)nd * DIM + n] = v;
          atomAddF(&pSum[g * DIM + n], v);             // per-graph node sum
        }
      }
    }
  }
  __syncthreads();
  for (int i = tid; i < N_GRAPH * DIM; i += 256)
    if (pSum[i] != 0.f) atomAddF(&nodesum[i], pSum[i]);
}

// ---------------------------------------------------------------------------
// Global MLP: 8 rows — plain f32 VALU, single block.
__global__ __launch_bounds__(256) void global_mlp_kernel(
    const float* __restrict__ nodesum, const float* __restrict__ edgesum,
    const float* __restrict__ ncnt, const float* __restrict__ ecnt,
    const float* __restrict__ uattr,
    const float* __restrict__ Wg1, const float* __restrict__ bg1,
    const float* __restrict__ Wg2, const float* __restrict__ bg2,
    float* __restrict__ gout) {
  __shared__ float gin[N_GRAPH * 384];
  __shared__ float hid[N_GRAPH * HID];
  const int tid = threadIdx.x;
  for (int i = tid; i < N_GRAPH * 384; i += 256) {
    const int g = i / 384, c = i % 384;
    float v;
    if (c < 128)       v = nodesum[g * DIM + c]         / fmaxf(ncnt[g], 1.f);
    else if (c < 256)  v = edgesum[g * DIM + (c - 128)] / fmaxf(ecnt[g], 1.f);
    else               v = uattr[g * DIM + (c - 256)];
    gin[i] = v;
  }
  __syncthreads();
  for (int i = tid; i < N_GRAPH * HID; i += 256) {
    const int g = i / HID, h = i % HID;
    float s = bg1[h];
    for (int k = 0; k < 384; ++k) s += gin[g * 384 + k] * Wg1[k * HID + h];
    hid[i] = fmaxf(s, 0.f);
  }
  __syncthreads();
  for (int i = tid; i < N_GRAPH * DIM; i += 256) {
    const int g = i / DIM, d = i % DIM;
    float s = bg2[d];
    for (int k = 0; k < HID; ++k) s += hid[g * HID + k] * Wg2[k * DIM + d];
    gout[i] = s;
  }
}

// ---------------------------------------------------------------------------
// Workspace layout (bytes). bf16 W^T blocks first (32B aligned), then fp32.
static constexpr size_t OFF_W1E  = 0;                                          // [256][512] bf16
static constexpr size_t OFF_W2E  = OFF_W1E + (size_t)HID * 512 * 2;            // [128][256] bf16
static constexpr size_t OFF_W1N  = OFF_W2E + (size_t)DIM * HID * 2;            // [256][384] bf16
static constexpr size_t OFF_W2N  = OFF_W1N + (size_t)HID * 384 * 2;            // [128][256] bf16
static constexpr size_t OFF_AGG  = OFF_W2N + (size_t)DIM * HID * 2;            // [N,128] f32
static constexpr size_t OFF_NSUM = OFF_AGG  + (size_t)N_NODES * DIM * 4;       // [8,128] f32
static constexpr size_t OFF_ESUM = OFF_NSUM + (size_t)N_GRAPH * DIM * 4;       // [8,128] f32
static constexpr size_t OFF_NCNT = OFF_ESUM + (size_t)N_GRAPH * DIM * 4;       // [8] f32
static constexpr size_t OFF_ECNT = OFF_NCNT + 8 * 4;                           // [8] f32

extern "C" void kernel_launch(void* const* d_in, const int* in_sizes, int n_in,
                              void* d_out, int out_size, void* d_ws, size_t ws_size,
                              hipStream_t stream) {
  (void)in_sizes; (void)n_in; (void)out_size; (void)ws_size;
  const float* node_attr = (const float*)d_in[0];
  const float* edge_attr = (const float*)d_in[1];
  const float* uattr     = (const float*)d_in[2];
  const int*   edge_idx2 = (const int*)d_in[3];     // [2, E]
  const int*   node_gid  = (const int*)d_in[4];
  const int*   edge_gid  = (const int*)d_in[5];
  const float* We1 = (const float*)d_in[6];  const float* be1 = (const float*)d_in[7];
  const float* We2 = (const float*)d_in[8];  const float* be2 = (const float*)d_in[9];
  const float* Wn1 = (const float*)d_in[10]; const float* bn1 = (const float*)d_in[11];
  const float* Wn2 = (const float*)d_in[12]; const float* bn2 = (const float*)d_in[13];
  const float* Wg1 = (const float*)d_in[14]; const float* bg1 = (const float*)d_in[15];
  const float* Wg2 = (const float*)d_in[16]; const float* bg2 = (const float*)d_in[17];

  const int* erow = edge_idx2;
  const int* ecol = edge_idx2 + N_EDGES;

  char* ws = (char*)d_ws;
  bf16_t* W1eT = (bf16_t*)(ws + OFF_W1E);
  bf16_t* W2eT = (bf16_t*)(ws + OFF_W2E);
  bf16_t* W1nT = (bf16_t*)(ws + OFF_W1N);
  bf16_t* W2nT = (bf16_t*)(ws + OFF_W2N);
  float*  agg     = (float*)(ws + OFF_AGG);
  float*  nodesum = (float*)(ws + OFF_NSUM);
  float*  edgesum = (float*)(ws + OFF_ESUM);
  float*  ncnt    = (float*)(ws + OFF_NCNT);
  float*  ecnt    = (float*)(ws + OFF_ECNT);

  float* node_out = (float*)d_out;
  float* edge_out = node_out + (size_t)N_NODES * DIM;
  float* g_out    = edge_out + (size_t)N_EDGES * DIM;

  // 1) zero agg + sums + counts (contiguous region starting at OFF_AGG)
  const long n_zero = (long)N_NODES * DIM + 2L * N_GRAPH * DIM + 16;
  zero_kernel<<<4096, 256, 0, stream>>>(agg, n_zero);

  // 2) weights -> bf16 transposed
  transpose_bf16_kernel<<<(512 * 256 + 255) / 256, 256, 0, stream>>>(We1, W1eT, 512, HID);
  transpose_bf16_kernel<<<(HID * DIM + 255) / 256, 256, 0, stream>>>(We2, W2eT, HID, DIM);
  transpose_bf16_kernel<<<(384 * 256 + 255) / 256, 256, 0, stream>>>(Wn1, W1nT, 384, HID);
  transpose_bf16_kernel<<<(HID * DIM + 255) / 256, 256, 0, stream>>>(Wn2, W2nT, HID, DIM);

  // 3) per-graph node/edge counts
  count_kernel<<<(N_EDGES + 255) / 256, 256, 0, stream>>>(node_gid, edge_gid, ncnt, ecnt);

  // 4) edge MLP (+ scatter to agg, per-graph edge sums)
  edge_mlp_kernel<<<(N_EDGES + MT - 1) / MT, 256, 0, stream>>>(
      node_attr, edge_attr, uattr, erow, ecol, edge_gid,
      W1eT, be1, W2eT, be2, edge_out, agg, edgesum);

  // 5) node MLP (+ per-graph node sums)
  node_mlp_kernel<<<(N_NODES + MT - 1) / MT, 256, 0, stream>>>(
      node_attr, agg, uattr, node_gid,
      W1nT, bn1, W2nT, bn2, node_out, nodesum);

  // 6) global MLP
  global_mlp_kernel<<<1, 256, 0, stream>>>(
      nodesum, edgesum, ncnt, ecnt, uattr, Wg1, bg1, Wg2, bg2, g_out);
}
